// HfLSHSelfAttention_27350351740988
// MI455X (gfx1250) — matrix operand order repaired and checked by reading, not compiled, not run on hardware
//
#include <hip/hip_runtime.h>
#include <hip/hip_bf16.h>

// ---------------------------------------------------------------------------
// LSH attention for MI455X (gfx1250), wave32 + WMMA bf16, packed LDS I/O,
// double-buffered GEMM w/ fused rotary, WMMA bucket projection, fast stable
// counting sort.   B=2, S=8192, H=1024, NH=16, D=64, CH=64, NHASH=2, NB=256
// ---------------------------------------------------------------------------

#define BB    2
#define SS    8192
#define HH    1024
#define NH    16
#define DD    64
#define CH    64
#define NHASH 2
#define NB    256
#define NSORT (NHASH * SS)     // 16384 per (b,h)
#define NCHUNK (NSORT / CH)    // 256
#define NBUCKET (NHASH * NB)   // 512

typedef __attribute__((ext_vector_type(16))) __bf16 v16bf;
typedef __attribute__((ext_vector_type(8)))  __bf16 v8bf;
typedef __attribute__((ext_vector_type(8)))  float  v8f;
typedef __attribute__((ext_vector_type(8)))  unsigned short u16x8;
typedef __attribute__((ext_vector_type(4)))  unsigned short u16x4;

// float -> bf16 (round-to-nearest-even) as raw u16 for LDS staging
__device__ __forceinline__ unsigned short stbf(float f) {
    unsigned u = __builtin_bit_cast(unsigned, f);
    return (unsigned short)((u + 0x7FFFu + ((u >> 16) & 1u)) >> 16);
}

// two 16B contiguous LDS chunks -> one 16-element bf16 WMMA fragment
__device__ __forceinline__ v16bf frag16(const unsigned short* lo,
                                        const unsigned short* hi) {
    v8bf l = __builtin_bit_cast(v8bf, *(const u16x8*)lo);
    v8bf h = __builtin_bit_cast(v8bf, *(const u16x8*)hi);
    return __builtin_shufflevector(l, h, 0, 1, 2, 3, 4, 5, 6, 7,
                                         8, 9, 10, 11, 12, 13, 14, 15);
}

#define WMMA_BF16(a, b, c) __builtin_amdgcn_wmma_f32_16x16x32_bf16( \
    false, (a), false, (b), (short)0, (c), false, false)

// argmax update with "first max wins" (matches jnp.argmax scan order)
__device__ __forceinline__ void amax_upd(float v, int idx, float& best, int& bi) {
    if (v > best || (v == best && idx < bi)) { best = v; bi = idx; }
}

// ---------------------------------------------------------------------------
// Stage one 128x32 A tile (row-major) + 32x128 B tile (transposed) as bf16.
// ---------------------------------------------------------------------------
__device__ __forceinline__ void stage_tiles(
    const float* __restrict__ X, const float* __restrict__ W,
    int bm, int bn, int k0, int tid,
    unsigned short (*As)[40], unsigned short (*Bst)[40]) {
    // A: thread -> 16 consecutive floats of one row
    {
        const int r = tid >> 1, seg = (tid & 1) * 16;
        const float* src = X + (size_t)(bm * 128 + r) * HH + k0 + seg;
        const float4 f0 = *(const float4*)(src + 0);
        const float4 f1 = *(const float4*)(src + 4);
        const float4 f2 = *(const float4*)(src + 8);
        const float4 f3 = *(const float4*)(src + 12);
        u16x8 p0, p1;
        p0[0] = stbf(f0.x); p0[1] = stbf(f0.y); p0[2] = stbf(f0.z); p0[3] = stbf(f0.w);
        p0[4] = stbf(f1.x); p0[5] = stbf(f1.y); p0[6] = stbf(f1.z); p0[7] = stbf(f1.w);
        p1[0] = stbf(f2.x); p1[1] = stbf(f2.y); p1[2] = stbf(f2.z); p1[3] = stbf(f2.w);
        p1[4] = stbf(f3.x); p1[5] = stbf(f3.y); p1[6] = stbf(f3.z); p1[7] = stbf(f3.w);
        *(u16x8*)&As[r][seg]     = p0;
        *(u16x8*)&As[r][seg + 8] = p1;
    }
    // B transposed: thread -> one column, 16 k values (coalesced across lanes)
    {
        const int col = tid & 127, kg = tid >> 7;      // kg in {0,1}
        const float* src = W + (size_t)(k0 + kg * 16) * 1024 + bn * 128 + col;
        u16x8 p0, p1;
#pragma unroll
        for (int j = 0; j < 8; ++j) {
            p0[j] = stbf(src[(size_t)j * 1024]);
            p1[j] = stbf(src[(size_t)(8 + j) * 1024]);
        }
        *(u16x8*)&Bst[col][kg * 16]     = p0;
        *(u16x8*)&Bst[col][kg * 16 + 8] = p1;
    }
}

// ---------------------------------------------------------------------------
// Kernel 1: Y = X @ W. Block 256 thr (8 waves), tile 128x128, K step 32,
// double-buffered LDS. do_rot: fuse interleaved rotary via shfl_xor(1).
// ---------------------------------------------------------------------------
__global__ __launch_bounds__(256) void gemm_bf16_wmma(
    const float* __restrict__ X,   // (M=B*S, 1024)
    const float* __restrict__ W,   // (1024, 1024)
    float* __restrict__ Y,         // ((b*NH+h)*S+s)*64 + d
    const int do_rot)
{
    __shared__ alignas(16) unsigned short As [2][128][40];
    __shared__ alignas(16) unsigned short Bst[2][128][40];   // [col][k]

    const int bm = blockIdx.x;          // 0..127
    const int bn = blockIdx.y;          // 0..7
    const int tid  = threadIdx.x;
    const int wave = tid >> 5;
    const int lane = tid & 31;
    const int mt = (wave & 3) * 32;
    const int ng = (wave >> 2) * 64;
    const int l15 = lane & 15;
    const int kh = (lane >> 4) * 8;
    const int kb = (lane >> 4) * 16;

    v8f acc[2][4] = {};

    stage_tiles(X, W, bm, bn, 0, tid, As[0], Bst[0]);
    __syncthreads();

    for (int k0 = 0; k0 < HH; k0 += 32) {
        const int cur = (k0 >> 5) & 1;
        if (k0 + 32 < HH) {
            stage_tiles(X, W, bm, bn, k0 + 32, tid, As[cur ^ 1], Bst[cur ^ 1]);
            if (k0 + 64 < HH) {   // gfx1250 global_prefetch for tile k0+64
                __builtin_prefetch(X + (size_t)(bm * 128 + (tid >> 1)) * HH + k0 + 64, 0, 1);
                __builtin_prefetch(W + (size_t)(k0 + 64 + (tid >> 7) * 16) * 1024 + bn * 128 + (tid & 127), 0, 1);
            }
        }
        v16bf a0 = frag16(&As[cur][mt +      l15][kh], &As[cur][mt +      l15][16 + kh]);
        v16bf a1 = frag16(&As[cur][mt + 16 + l15][kh], &As[cur][mt + 16 + l15][16 + kh]);
#pragma unroll
        for (int nt = 0; nt < 4; ++nt) {
            const int bc = ng + nt * 16 + l15;
            v16bf b = frag16(&Bst[cur][bc][kh], &Bst[cur][bc][16 + kh]);
            acc[0][nt] = WMMA_BF16(a0, b, acc[0][nt]);
            acc[1][nt] = WMMA_BF16(a1, b, acc[1][nt]);
        }
        __syncthreads();
    }

#pragma unroll
    for (int ms = 0; ms < 2; ++ms) {
#pragma unroll
        for (int nt = 0; nt < 4; ++nt) {
            const int col = bn * 128 + ng + nt * 16 + l15;
            const int rb  = bm * 128 + mt + ms * 16 + (lane >> 4) * 8;
            const int h = col >> 6, d = col & 63;
            // pair index exponent: 10000^(-(2i)/64), i = d>>1 -> (d&62)/64
            const float invf = __powf(10000.0f, -(float)(d & 62) / 64.0f);
#pragma unroll
            for (int r = 0; r < 8; ++r) {
                const int row = rb + r;                 // m = b*S + s
                const int b = row >> 13, s = row & (SS - 1);
                float val = acc[ms][nt][r];
                const float other = __shfl_xor(val, 1); // pair lives in lane^1
                if (do_rot) {
                    const float ang = (float)s * invf;
                    const float cc = __cosf(ang), sn = __sinf(ang);
                    val = (d & 1) ? (val * cc + other * sn)
                                  : (val * cc - other * sn);
                }
                Y[((((size_t)b * NH + h) * SS) + s) * DD + d] = val;
            }
        }
    }
}

// ---------------------------------------------------------------------------
// Kernel 2a: transpose rotations (NH,D,NHASH,128) -> (NH, NHASH*128, D)
// ---------------------------------------------------------------------------
__global__ void rott_kernel(const float* __restrict__ rot,
                            float* __restrict__ rott) {
    const int total = NH * NHASH * 128 * DD;    // 262144
    const int gid = blockIdx.x * blockDim.x + threadIdx.x;
    if (gid >= total) return;
    const int d = gid & 63;
    int rem = gid >> 6;
    const int r = rem & 127; rem >>= 7;
    const int n = rem & 1;
    const int h = rem >> 1;
    rott[((((size_t)h * NHASH + n) * 128) + r) * DD + d] =
        rot[(((size_t)h * DD + d) * NHASH + n) * 128 + r];
}

// ---------------------------------------------------------------------------
// Kernel 2b: WMMA LSH bucket projection. Block = 128 thr (4 waves) per
// (bh, 64-row s-tile). dots(64x256) = qk_tile(64x64) @ rott_head^T, then
// per-(row,hash) argmax over [dots, -dots] with first-max tie rule.
// ---------------------------------------------------------------------------
__global__ __launch_bounds__(128) void bucket_wmma_kernel(
    const float* __restrict__ qk,
    const float* __restrict__ rott,    // (NH, 256, 64)
    unsigned int* __restrict__ bucket) {
    __shared__ alignas(16) unsigned short q_bf[64][72];
    __shared__ alignas(16) unsigned short r_bf[256][72];
    __shared__ alignas(16) float dots[64][260];

    const int blk = blockIdx.x;        // bh*128 + stile
    const int st = blk & 127;
    const int bh = blk >> 7;
    const int h  = bh & 15;
    const int tid = threadIdx.x, wave = tid >> 5, lane = tid & 31;
    const int lsub = tid & 15, rgrp = tid >> 4;

    const float* qb = qk + (size_t)bh * SS * DD + (size_t)st * 64 * DD;
    const float* rbase = rott + (size_t)h * 256 * DD;

    for (int rr = 0; rr < 64; rr += 8) {          // stage 64 query rows
        const int row = rr + rgrp;
        const float4 f = *(const float4*)(qb + (size_t)row * DD + lsub * 4);
        u16x4 p; p[0] = stbf(f.x); p[1] = stbf(f.y); p[2] = stbf(f.z); p[3] = stbf(f.w);
        *(u16x4*)&q_bf[row][lsub * 4] = p;
    }
    for (int rr = 0; rr < 256; rr += 8) {         // stage 256 rotation rows
        const int row = rr + rgrp;
        const float4 f = *(const float4*)(rbase + (size_t)row * DD + lsub * 4);
        u16x4 p; p[0] = stbf(f.x); p[1] = stbf(f.y); p[2] = stbf(f.z); p[3] = stbf(f.w);
        *(u16x4*)&r_bf[row][lsub * 4] = p;
    }
    __syncthreads();

    const int l15 = lane & 15;
    const int kh = (lane >> 4) * 8;
    const int kb = (lane >> 4) * 16;
    const int mrow = wave * 16 + l15;
    const int rbr = wave * 16 + (lane >> 4) * 8;
    {
        v8f acc[16] = {};
#pragma unroll
        for (int kk = 0; kk < 64; kk += 32) {
            v16bf a = frag16(&q_bf[mrow][kk + kh], &q_bf[mrow][kk + 16 + kh]);
#pragma unroll
            for (int nt = 0; nt < 16; ++nt) {
                const int bn = nt * 16 + l15;
                v16bf b = frag16(&r_bf[bn][kk + kh], &r_bf[bn][kk + 16 + kh]);
                acc[nt] = WMMA_BF16(a, b, acc[nt]);
            }
        }
#pragma unroll
        for (int nt = 0; nt < 16; ++nt) {
            const int col = nt * 16 + l15;
#pragma unroll
            for (int r = 0; r < 8; ++r) dots[rbr + r][col] = acc[nt][r];
        }
    }
    __syncthreads();

    // argmax: thread t handles (row = t&63, hash n = t>>6) over 128 cols
    {
        const int row = tid & 63, n = tid >> 6;
        const float* dp = &dots[row][n * 128];
        float best = -3.4e38f; int bi = 0;
        for (int r = 0; r < 128; r += 4) {
            const float4 d4 = *(const float4*)(dp + r);
            amax_upd( d4.x, r + 0,       best, bi);
            amax_upd(-d4.x, r + 0 + 128, best, bi);
            amax_upd( d4.y, r + 1,       best, bi);
            amax_upd(-d4.y, r + 1 + 128, best, bi);
            amax_upd( d4.z, r + 2,       best, bi);
            amax_upd(-d4.z, r + 2 + 128, best, bi);
            amax_upd( d4.w, r + 3,       best, bi);
            amax_upd(-d4.w, r + 3 + 128, best, bi);
        }
        const int t = st * 64 + row;
        bucket[(size_t)bh * NSORT + n * SS + t] = (unsigned)(bi + n * NB);
    }
}

// ---------------------------------------------------------------------------
// Kernel 3: stable chunked counting sort per (b,h).
// 32 chunks x 512 elements; per-chunk histograms in LDS; bucket-major scan;
// stable per-chunk scatter. Serial critical path ~512 iterations.
// ---------------------------------------------------------------------------
__global__ __launch_bounds__(512) void sort_kernel(
    const unsigned int* __restrict__ bucket,
    int* __restrict__ sidx, int* __restrict__ undo) {
    __shared__ unsigned int cnt[NBUCKET][33];    // [bucket][chunk], padded
    __shared__ unsigned int bucket_base[NBUCKET];
    const int bh = blockIdx.x;
    const unsigned int* bk = bucket + (size_t)bh * NSORT;
    const int tid = threadIdx.x;

    {   // zero
        unsigned int* flat = &cnt[0][0];
        for (int i = tid; i < NBUCKET * 33; i += 512) flat[i] = 0;
    }
    __syncthreads();

    // phase 1: per-chunk histograms (sequential within chunk => stable)
    if (tid < 32) {
        const unsigned int* p = bk + tid * 512;
        for (int j = 0; j < 512; ++j) cnt[p[j]][tid]++;
    }
    __syncthreads();

    // phase 2a: per-bucket exclusive prefix over chunks; row total out
    {
        unsigned int acc = 0;
#pragma unroll 4
        for (int ch = 0; ch < 32; ++ch) {
            const unsigned int t = cnt[tid][ch];
            cnt[tid][ch] = acc;
            acc += t;
        }
        bucket_base[tid] = acc;
    }
    __syncthreads();

    // phase 2b: serial exclusive scan of 512 bucket totals
    if (tid == 0) {
        unsigned int acc = 0;
        for (int b2 = 0; b2 < NBUCKET; ++b2) {
            const unsigned int t = bucket_base[b2];
            bucket_base[b2] = acc;
            acc += t;
        }
    }
    __syncthreads();

    // phase 2c: add bucket base into this bucket's chunk offsets
    {
        const unsigned int base = bucket_base[tid];
#pragma unroll 4
        for (int ch = 0; ch < 32; ++ch) cnt[tid][ch] += base;
    }
    __syncthreads();

    // phase 3: stable scatter (sequential within chunk)
    if (tid < 32) {
        const unsigned int* p = bk + tid * 512;
        int* so = sidx + (size_t)bh * NSORT;
        int* un = undo + (size_t)bh * NSORT;
        for (int j = 0; j < 512; ++j) {
            const unsigned int bb = p[j];
            const int pos = (int)cnt[bb][tid]++;
            const int src = tid * 512 + j;
            so[pos] = src;
            un[src] = pos;
        }
    }
}

// ---------------------------------------------------------------------------
// Kernel 4: chunked attention. 1 workgroup (128 thr / 4 waves) per
// (b,h,chunk). 16 lanes cooperate per gathered row; V staged transposed so
// both WMMA stages use contiguous two-b128 fragments.
// ---------------------------------------------------------------------------
__global__ __launch_bounds__(128) void attn_kernel(
    const float* __restrict__ qk, const float* __restrict__ v,
    const int* __restrict__ sidx,
    float* __restrict__ outb,      // (bh, 16384, 64)
    float* __restrict__ logitsb)   // (bh, 16384)
{
    __shared__ alignas(16) unsigned short q_bf[64][72];    // [qrow][d]
    __shared__ alignas(16) unsigned short k_bf[128][72];   // [krow][d] (normalized)
    __shared__ alignas(16) unsigned short v_t [64][136];   // [d][krow]
    __shared__ alignas(16) unsigned short p_bf[64][136];   // [qrow][k]
    __shared__ alignas(16) float dots[64][132];
    __shared__ alignas(16) int   qi[64];
    __shared__ alignas(16) int   ki[128];
    __shared__ alignas(16) float rowinv[64];

    const int blk = blockIdx.x;
    const int c  = blk & (NCHUNK - 1);
    const int bh = blk >> 8;
    const int pc = (c + NCHUNK - 1) & (NCHUNK - 1);   // jnp.roll wrap
    const int tid = threadIdx.x, wave = tid >> 5, lane = tid & 31;
    const int lsub = tid & 15, rgrp = tid >> 4;       // 16 lanes per row, 8 rows

    const int*   sx  = sidx + (size_t)bh * NSORT;
    const float* qkb = qk + (size_t)bh * SS * DD;
    const float* vb  = v  + (size_t)bh * SS * DD;

    // ---- gather queries (raw rotary'd qk rows)
    for (int rr = 0; rr < 64; rr += 8) {
        const int row = rr + rgrp;
        const int t = sx[c * CH + row] & (SS - 1);
        if (lsub == 0) qi[row] = t;
        const float4 f = *(const float4*)(qkb + (size_t)t * DD + lsub * 4);
        u16x4 p; p[0] = stbf(f.x); p[1] = stbf(f.y); p[2] = stbf(f.z); p[3] = stbf(f.w);
        *(u16x4*)&q_bf[row][lsub * 4] = p;
    }
    // ---- gather keys (normalized) + values (transposed store)
    for (int rr = 0; rr < 128; rr += 8) {
        const int row = rr + rgrp;
        const int cc = (row < 64) ? pc : c;
        const int r  = (row < 64) ? row : row - 64;
        const int t  = sx[cc * CH + r] & (SS - 1);
        if (lsub == 0) ki[row] = t;
        const float4 f = *(const float4*)(qkb + (size_t)t * DD + lsub * 4);
        float ss = f.x * f.x + f.y * f.y + f.z * f.z + f.w * f.w;
#pragma unroll
        for (int off = 1; off < 16; off <<= 1) ss += __shfl_xor(ss, off);
        const float scale = rsqrtf(ss * (1.0f / DD) + 1e-6f) * 0.125f; // 1/sqrt(64)
        u16x4 p;
        p[0] = stbf(f.x * scale); p[1] = stbf(f.y * scale);
        p[2] = stbf(f.z * scale); p[3] = stbf(f.w * scale);
        *(u16x4*)&k_bf[row][lsub * 4] = p;
        const float4 g = *(const float4*)(vb + (size_t)t * DD + lsub * 4);
        v_t[lsub * 4 + 0][row] = stbf(g.x);
        v_t[lsub * 4 + 1][row] = stbf(g.y);
        v_t[lsub * 4 + 2][row] = stbf(g.z);
        v_t[lsub * 4 + 3][row] = stbf(g.w);
    }
    __syncthreads();

    const int l15 = lane & 15;
    const int kh = (lane >> 4) * 8;
    const int kb = (lane >> 4) * 16;
    const int mrow = wave * 16 + l15;
    const int rb = wave * 16 + (lane >> 4) * 8;

    // ---- dots(64x128) = Q(64x64) . K^T
    {
        v8f acc[8] = {};
#pragma unroll
        for (int kk = 0; kk < 64; kk += 32) {
            v16bf a = frag16(&q_bf[mrow][kk + kh], &q_bf[mrow][kk + 16 + kh]);
#pragma unroll
            for (int nt = 0; nt < 8; ++nt) {
                const int bn = nt * 16 + l15;
                v16bf b = frag16(&k_bf[bn][kk + kh], &k_bf[bn][kk + 16 + kh]);
                acc[nt] = WMMA_BF16(a, b, acc[nt]);
            }
        }
#pragma unroll
        for (int nt = 0; nt < 8; ++nt) {
            const int col = nt * 16 + l15;
#pragma unroll
            for (int r = 0; r < 8; ++r) dots[rb + r][col] = acc[nt][r];
        }
    }
    __syncthreads();

    // ---- masks + logsumexp (thread per query row, vectorized reads)
    for (int row = tid; row < 64; row += 128) {
        const int qq = qi[row];
        float mx = -3.4e38f;
        for (int k2 = 0; k2 < 128; k2 += 4) {
            const float4 d4 = *(const float4*)&dots[row][k2];
            const int4   k4 = *(const int4*)&ki[k2];
            float m0 = (qq == k4.x) ? -1.0e5f : ((qq < k4.x) ? -1.0e9f : d4.x);
            float m1 = (qq == k4.y) ? -1.0e5f : ((qq < k4.y) ? -1.0e9f : d4.y);
            float m2 = (qq == k4.z) ? -1.0e5f : ((qq < k4.z) ? -1.0e9f : d4.z);
            float m3 = (qq == k4.w) ? -1.0e5f : ((qq < k4.w) ? -1.0e9f : d4.w);
            mx = fmaxf(mx, fmaxf(fmaxf(m0, m1), fmaxf(m2, m3)));
        }
        float sum = 0.0f;
        for (int k2 = 0; k2 < 128; k2 += 4) {
            const float4 d4 = *(const float4*)&dots[row][k2];
            const int4   k4 = *(const int4*)&ki[k2];
            float m0 = (qq == k4.x) ? -1.0e5f : ((qq < k4.x) ? -1.0e9f : d4.x);
            float m1 = (qq == k4.y) ? -1.0e5f : ((qq < k4.y) ? -1.0e9f : d4.y);
            float m2 = (qq == k4.z) ? -1.0e5f : ((qq < k4.z) ? -1.0e9f : d4.z);
            float m3 = (qq == k4.w) ? -1.0e5f : ((qq < k4.w) ? -1.0e9f : d4.w);
            const float e0 = __expf(m0 - mx), e1 = __expf(m1 - mx);
            const float e2 = __expf(m2 - mx), e3 = __expf(m3 - mx);
            sum += (e0 + e1) + (e2 + e3);
            u16x4 p; p[0] = stbf(e0); p[1] = stbf(e1); p[2] = stbf(e2); p[3] = stbf(e3);
            *(u16x4*)&p_bf[row][k2] = p;     // 1/sum folded in epilogue
        }
        rowinv[row] = 1.0f / sum;
        logitsb[(size_t)bh * NSORT + c * CH + row] = __logf(sum) + mx;
    }
    __syncthreads();

    // ---- out(64x64) = P(64x128) . V(128x64), V pre-transposed
    {
        v8f acc[4] = {};
#pragma unroll
        for (int kk = 0; kk < 128; kk += 32) {
            v16bf a = frag16(&p_bf[mrow][kk + kh], &p_bf[mrow][kk + 16 + kh]);
#pragma unroll
            for (int nt = 0; nt < 4; ++nt) {
                const int bn = nt * 16 + l15;
                v16bf b = frag16(&v_t[bn][kk + kh], &v_t[bn][kk + 16 + kh]);
                acc[nt] = WMMA_BF16(a, b, acc[nt]);
            }
        }
        float* ob = outb + ((size_t)bh * NSORT + (size_t)c * CH) * DD;
#pragma unroll
        for (int nt = 0; nt < 4; ++nt) {
            const int col = nt * 16 + l15;
#pragma unroll
            for (int r = 0; r < 8; ++r) {
                const int row = rb + r;
                ob[(size_t)row * DD + col] = acc[nt][r] * rowinv[row];
            }
        }
    }
}

// ---------------------------------------------------------------------------
// Kernel 5: combine the 2 hash rounds with softmax(logits) weights, unsort.
// ---------------------------------------------------------------------------
__global__ void combine_kernel(const float* __restrict__ outb,
                               const float* __restrict__ logitsb,
                               const int* __restrict__ undo,
                               float* __restrict__ out) {
    const int total = BB * NH * SS * (DD / 4);  // 4,194,304
    const int gid = blockIdx.x * blockDim.x + threadIdx.x;
    if (gid >= total) return;
    const int d4 = gid & 15;
    const int row = gid >> 4;              // (b*NH+h)*S + s
    const int s = row & (SS - 1);
    const int bh = row >> 13;
    const int h = bh & 15, b = bh >> 4;

    const int* un = undo + (size_t)bh * NSORT;
    const int p0 = un[s], p1 = un[SS + s];
    const float l0 = logitsb[(size_t)bh * NSORT + p0];
    const float l1 = logitsb[(size_t)bh * NSORT + p1];
    const float m = fmaxf(l0, l1);
    const float w0 = __expf(l0 - m), w1 = __expf(l1 - m);
    const float inv = 1.0f / (w0 + w1);
    const float* ob = outb + (size_t)bh * NSORT * DD;
    const float4 a = *(const float4*)(ob + (size_t)p0 * DD + d4 * 4);
    const float4 bq = *(const float4*)(ob + (size_t)p1 * DD + d4 * 4);
    float4 o;
    o.x = (w0 * a.x + w1 * bq.x) * inv;
    o.y = (w0 * a.y + w1 * bq.y) * inv;
    o.z = (w0 * a.z + w1 * bq.z) * inv;
    o.w = (w0 * a.w + w1 * bq.w) * inv;
    *(float4*)(out + (((size_t)b * SS + s) * (NH * DD)) + h * DD + d4 * 4) = o;
}

// ---------------------------------------------------------------------------
extern "C" void kernel_launch(void* const* d_in, const int* in_sizes, int n_in,
                              void* d_out, int out_size, void* d_ws, size_t ws_size,
                              hipStream_t stream) {
    (void)in_sizes; (void)n_in; (void)out_size; (void)ws_size;
    const float* hidden = (const float*)d_in[0];  // (2, 8192, 1024)
    const float* w_qk   = (const float*)d_in[1];  // (1024, 1024)
    const float* w_v    = (const float*)d_in[2];  // (1024, 1024)
    const float* rots   = (const float*)d_in[3];  // (16, 64, 2, 128)
    float* out = (float*)d_out;

    // workspace layout (bytes)
    char* ws = (char*)d_ws;
    const size_t qk_bytes   = (size_t)BB * NH * SS * DD * 4;    // 64 MB
    const size_t v_bytes    = qk_bytes;                         // 64 MB
    const size_t bkt_bytes  = (size_t)BB * NH * NSORT * 4;      // 2 MB
    const size_t idx_bytes  = bkt_bytes;                        // 2 MB
    const size_t outb_bytes = (size_t)BB * NH * NSORT * DD * 4; // 128 MB
    const size_t log_bytes  = (size_t)BB * NH * NSORT * 4;      // 2 MB
    size_t off = 0;
    float* qk_ws   = (float*)(ws + off); off += qk_bytes;
    float* v_ws    = (float*)(ws + off); off += v_bytes;
    unsigned int* bucket_ws = (unsigned int*)(ws + off); off += bkt_bytes;
    int* sidx_ws   = (int*)(ws + off); off += idx_bytes;
    int* undo_ws   = (int*)(ws + off); off += idx_bytes;
    float* outb_ws = (float*)(ws + off); off += outb_bytes;
    float* logit_ws = (float*)(ws + off); off += log_bytes;
    float* rott_ws = (float*)(ws + off); off += (size_t)NH * NHASH * 128 * DD * 4;

    // 0) rotations transpose (independent of GEMMs)
    rott_kernel<<<(NH * NHASH * 128 * DD) / 256, 256, 0, stream>>>(rots, rott_ws);

    // 1+2) projections via WMMA (rotary fused into the qk epilogue)
    dim3 ggrid(BB * SS / 128, 1024 / 128, 1);
    gemm_bf16_wmma<<<ggrid, 256, 0, stream>>>(hidden, w_qk, qk_ws, 1);
    gemm_bf16_wmma<<<ggrid, 256, 0, stream>>>(hidden, w_v,  v_ws, 0);

    // 3) LSH buckets via WMMA projection
    bucket_wmma_kernel<<<BB * NH * 128, 128, 0, stream>>>(qk_ws, rott_ws, bucket_ws);

    // 4) stable chunked counting sort per (b,h)
    sort_kernel<<<BB * NH, 512, 0, stream>>>(bucket_ws, sidx_ws, undo_ws);

    // 5) chunked attention (WMMA)
    attn_kernel<<<BB * NH * NCHUNK, 128, 0, stream>>>(qk_ws, v_ws, sidx_ws, outb_ws, logit_ws);

    // 6) combine hash rounds + unsort
    {
        const int total = BB * NH * SS * (DD / 4); // 4.2M
        combine_kernel<<<total / 256, 256, 0, stream>>>(outb_ws, logit_ws, undo_ws, out);
    }
}
